// model_11175504904743
// MI455X (gfx1250) — matrix operand order repaired and checked by reading, not compiled, not run on hardware
//
#include <hip/hip_runtime.h>
#include <hip/hip_bf16.h>

// ---------------------------------------------------------------------------
// Constants from the reference
// ---------------------------------------------------------------------------
#define KTOP        100
#define NUM_CLASSES 80
#define KEEP        300
#define IOU_TH      0.5f
#define ROI_HW      9          // 2*MAX+1
#define MASK_OUT    36
#define SCALE_C     4.375f     // (36-1)/(9-1)
#define BN_INV_C    0.9995003746877732f  // 1/sqrt(1.001)

typedef __attribute__((ext_vector_type(16))) __bf16 v16bf;
typedef __attribute__((ext_vector_type(8)))  float  v8f;

union FragBF { v16bf v; unsigned u[8]; unsigned short us[16]; };

__device__ __forceinline__ unsigned short f2bf(float f) {
  unsigned u = __float_as_uint(f);
  u += 0x7fffu + ((u >> 16) & 1u);     // round-to-nearest-even
  return (unsigned short)(u >> 16);
}

struct GemmDesc {
  int M, K, Cin, Hin, Win, Hout, Wout, KH, KW, pad;
  int Npix;      // N dimension: conv -> Hout*Wout, deconv -> Hin*Win
  int mode;      // 0 = conv (im2col), 1 = deconv (per-parity-class GEMM)
  int act;       // 0 none, 1 relu, 2 sigmoid
  int outF32;    // 1 -> write f32, else bf16
  long inStride, outStride;   // per-ROI element strides
};

// ---------------------------------------------------------------------------
// Weight pre-pack kernels (f32 -> bf16)
// ---------------------------------------------------------------------------
__global__ void pk_f32_to_bf16(const float* __restrict__ src,
                               unsigned short* __restrict__ dst, int n) {
  int i = blockIdx.x * 256 + threadIdx.x;
  if (i < n) dst[i] = f2bf(src[i]);
}

// deconv weight (O=256,I=256,2,2) -> [class=ky*2+kx][O][I] bf16
__global__ void pk_deconv(const float* __restrict__ src,
                          unsigned short* __restrict__ dst) {
  int i = blockIdx.x * 256 + threadIdx.x;
  if (i < 4 * 256 * 256) {
    int c = i >> 16, o = (i >> 8) & 255, ii = i & 255;
    dst[i] = f2bf(src[(o * 256 + ii) * 4 + c]);
  }
}

// ---------------------------------------------------------------------------
// Generic WMMA implicit-GEMM conv / deconv kernel
//   grid = (nRoi, nNtiles, nClasses)   block = 256 (8 wave32)
//
// The block first builds a packed im2col B-panel in LDS:
//   bp[octet = K/16][col = 16][slot = 8]  (uint = bf16 pair (2k, 2k+1))
// so that each lane's 8-dword B fragment for one WMMA step is 32 contiguous
// bytes (two ds_load_b128). The A fragment (contiguous k order, k = 16h+s,
// as in B) is exactly two contiguous 16B runs per lane -> two
// global_load_b128. The K-loop is then 2 VMEM + 2 DS + 1 WMMA.
// ---------------------------------------------------------------------------
__global__ __launch_bounds__(256) void wmma_gemm_kernel(
    const unsigned short* __restrict__ A,     // bf16 weights, row-major M x K (x class)
    const unsigned short* __restrict__ inAct, // bf16 per-ROI activations [C][H][W]
    void* __restrict__ outAct,
    const float* __restrict__ bias,
    const float* __restrict__ bnG,            // may be null
    const float* __restrict__ bnB,
    GemmDesc d) {
  extern __shared__ unsigned bp[];           // K*2 bytes per column block
  const int roi = blockIdx.x;
  const int nt  = blockIdx.y;
  const int cls = blockIdx.z;               // deconv parity class
  const int tid = threadIdx.x;
  const int lid = tid & 31;
  const int wid = tid >> 5;

  const int Wpix = (d.mode == 1) ? d.Win : d.Wout;
  const int P0 = nt * 16;

  // per-thread column (invariant: 256 % 16 == 0)
  const int col = tid & 15;
  const int P   = min(P0 + col, d.Npix - 1);
  const int py  = P / Wpix, px = P - py * Wpix;

  // ---- cooperative build of the im2col B panel --------------------------
  const unsigned short* src = inAct + (long)roi * d.inStride;
  const int HWin   = d.Hin * d.Win;
  const int nElem  = (d.K >> 1) * 16;       // (pair, col) pairs
  if (d.KH == 3) {
    // 3x3 conv, pad == 1 (constant divisors -> mul/shift)
    for (int idx = tid; idx < nElem; idx += 256) {
      int p  = idx >> 4;
      int k0 = p << 1;
      unsigned short e[2];
#pragma unroll
      for (int t = 0; t < 2; ++t) {
        int k  = k0 + t;
        int ci = k / 9;  int r  = k - ci * 9;
        int ky = r / 3;  int kx = r - ky * 3;
        int iy = py + ky - 1, ix = px + kx - 1;
        e[t] = (iy < 0 || iy >= d.Hin || ix < 0 || ix >= d.Win)
                 ? (unsigned short)0 : src[ci * HWin + iy * d.Win + ix];
      }
      int o = p >> 3, slot = p & 7;
      bp[((o * 16) + col) * 8 + slot] = (unsigned)e[0] | ((unsigned)e[1] << 16);
    }
  } else {
    // 1x1 conv (pad 0) or deconv: k == ci, pixel is (py, px) directly
    const int base = py * d.Win + px;
    for (int idx = tid; idx < nElem; idx += 256) {
      int p  = idx >> 4;
      int k0 = p << 1;
      unsigned short e0 = src[k0 * HWin + base];
      unsigned short e1 = src[(k0 + 1) * HWin + base];
      int o = p >> 3, slot = p & 7;
      bp[((o * 16) + col) * 8 + slot] = (unsigned)e0 | ((unsigned)e1 << 16);
    }
  }
  __syncthreads();

  const int aHalf = (lid >= 16) ? 16 : 0;    // A K-offset for hi lanes
  const int bSel  = (lid >= 16) ? 1 : 0;     // B octet select for hi lanes
  const unsigned short* Abase = A + (long)cls * d.M * d.K;

  for (int mt = wid; mt * 16 < d.M; mt += 8) {
    const unsigned short* Arow =
        Abase + (long)(mt * 16 + (lid & 15)) * d.K + aHalf;
    v8f acc = {0.f, 0.f, 0.f, 0.f, 0.f, 0.f, 0.f, 0.f};

    for (int k0 = 0; k0 < d.K; k0 += 32) {
      FragBF fa, fb;
      // A: one contiguous 32B run: k0+aHalf..+15
      const uint4* ap = (const uint4*)(Arow + k0);
      uint4 a0 = ap[0];
      uint4 a1 = ap[1];
      fa.u[0] = a0.x; fa.u[1] = a0.y; fa.u[2] = a0.z; fa.u[3] = a0.w;
      fa.u[4] = a1.x; fa.u[5] = a1.y; fa.u[6] = a1.z; fa.u[7] = a1.w;
      // B: 8 consecutive pair-dwords for this lane's octet
      const uint4* bq =
          (const uint4*)&bp[(((k0 >> 4) + bSel) * 16 + col) * 8];
      uint4 b0 = bq[0];
      uint4 b1 = bq[1];
      fb.u[0] = b0.x; fb.u[1] = b0.y; fb.u[2] = b0.z; fb.u[3] = b0.w;
      fb.u[4] = b1.x; fb.u[5] = b1.y; fb.u[6] = b1.z; fb.u[7] = b1.w;
      acc = __builtin_amdgcn_wmma_f32_16x16x32_bf16(
          false, fa.v, false, fb.v, (short)0, acc, false, false);
    }

    // epilogue: D layout -> VGPR r holds M = r (lanes 0-15) / r+8 (lanes 16-31)
    if (P0 + col < d.Npix) {
      const int mBase = mt * 16 + ((lid >= 16) ? 8 : 0);
#pragma unroll
      for (int r = 0; r < 8; ++r) {
        int m = mBase + r;
        float val = acc[r] + bias[m];
        if (bnG) val = val * (bnG[m] * BN_INV_C) + bnB[m];
        if (d.act == 1)      val = fmaxf(val, 0.f);
        else if (d.act == 2) val = 1.f / (1.f + expf(-val));
        long oidx;
        if (d.mode == 1) {
          int ky = cls >> 1, kx = cls & 1;
          int oy = 2 * py + (1 - ky), ox = 2 * px + (1 - kx);
          oidx = (long)roi * d.outStride + (long)m * (d.Hout * d.Wout)
               + oy * d.Wout + ox;
        } else {
          oidx = (long)roi * d.outStride + (long)m * d.Npix + P;
        }
        if (d.outF32) ((float*)outAct)[oidx] = val;
        else          ((unsigned short*)outAct)[oidx] = f2bf(val);
      }
    }
  }
}

// ---------------------------------------------------------------------------
// top-K (exact, sorted, stable) per pyramid layer — iterative lexicographic argmax
// ---------------------------------------------------------------------------
__global__ __launch_bounds__(1024) void topk_kernel(
    const float* __restrict__ h0, const float* __restrict__ h1,
    const float* __restrict__ h2, float* __restrict__ topVal,
    int* __restrict__ topIdx) {
  __shared__ float sv[1024];
  __shared__ int   si[1024];
  __shared__ float s_pv; __shared__ int s_pi;
  const int layer = blockIdx.x;
  const float* h = (layer == 0) ? h0 : (layer == 1) ? h1 : h2;
  const int hw = (layer == 0) ? 4096 : (layer == 1) ? 1024 : 256;
  const int N  = NUM_CLASSES * hw;
  const int tid = threadIdx.x;
  if (tid == 0) { s_pv = 3.4e38f; s_pi = -1; }
  __syncthreads();
  for (int r = 0; r < KTOP; ++r) {
    float pv = s_pv; int pi = s_pi;
    float bv = -3.4e38f; int bi = 0x7fffffff;
    for (int i = tid; i < N; i += 1024) {
      float v = h[i];
      if (v < pv || (v == pv && i > pi)) {          // strictly "after" previous pick
        if (v > bv || (v == bv && i < bi)) { bv = v; bi = i; }
      }
    }
    sv[tid] = bv; si[tid] = bi;
    __syncthreads();
    for (int s = 512; s > 0; s >>= 1) {
      if (tid < s) {
        float v2 = sv[tid + s]; int i2 = si[tid + s];
        if (v2 > sv[tid] || (v2 == sv[tid] && i2 < si[tid])) { sv[tid] = v2; si[tid] = i2; }
      }
      __syncthreads();
    }
    if (tid == 0) {
      topVal[layer * KTOP + r] = sv[0];
      topIdx[layer * KTOP + r] = si[0];
      s_pv = sv[0]; s_pi = si[0];
    }
    __syncthreads();
  }
}

// ---------------------------------------------------------------------------
// box decode + intra-layer stable re-rank (the reference's second top_k(sc,K))
// ---------------------------------------------------------------------------
__global__ __launch_bounds__(128) void decode_kernel(
    const float* tl0, const float* tl1, const float* tl2,
    const float* br0, const float* br1, const float* br2,
    const float* __restrict__ topVal, const int* __restrict__ topIdx,
    float* __restrict__ dets, float* __restrict__ bws, float* __restrict__ mb) {
  __shared__ float s_sc[KTOP];
  const int layer = blockIdx.x;
  const int j = threadIdx.x;
  const int W = 64 >> layer, H = 64 >> layer, hw = H * W;
  const float* tl = (layer == 0) ? tl0 : (layer == 1) ? tl1 : tl2;
  const float* br = (layer == 0) ? br0 : (layer == 1) ? br1 : br2;

  float tlx = 0, tly = 0, brx = 0, bry = 0, scn = 0, xs = 0, ys = 0, clsf = 0;
  if (j < KTOP) {
    int ind = topIdx[layer * KTOP + j];
    float sc = topVal[layer * KTOP + j];
    int clsI = ind / hw, rem = ind - clsI * hw;
    ys = (float)(rem / W); xs = (float)(rem % W); clsf = (float)clsI;
    float t0 = tl[rem], t1 = tl[hw + rem];
    float b0 = br[rem], b1 = br[hw + rem];
    tlx = fminf(fmaxf(xs - (3.f * t0 + 2.5f), 0.f), (float)W);
    tly = fminf(fmaxf(ys - (3.f * t1 + 2.5f), 0.f), (float)H);
    brx = fminf(fmaxf(xs + (3.f * b0 + 2.5f), 0.f), (float)W);
    bry = fminf(fmaxf(ys + (3.f * b1 + 2.5f), 0.f), (float)H);
    scn = ((brx < tlx) || (bry < tly)) ? -1.f : sc;
    s_sc[j] = scn;
  }
  __syncthreads();
  if (j < KTOP) {
    int rank = 0;
    for (int q = 0; q < KTOP; ++q) {
      float o = s_sc[q];
      if (o > scn || (o == scn && q < j)) rank++;
    }
    int row = layer * KTOP + rank;
    float sx = 64.f / (float)W, sy = 64.f / (float)H;
    float lay = (float)layer;
    float* dr = dets + row * 8;
    dr[0] = tlx * sx; dr[1] = tly * sy; dr[2] = brx * sx; dr[3] = bry * sy;
    dr[4] = scn; dr[5] = scn; dr[6] = lay; dr[7] = clsf;
    float* br_ = bws + row * 8;
    br_[0] = tlx; br_[1] = tly; br_[2] = brx; br_[3] = bry;
    br_[4] = scn; br_[5] = scn; br_[6] = lay; br_[7] = clsf;
    float* mr = mb + row * 5;
    mr[0] = xs - 4.f; mr[1] = ys - 4.f; mr[2] = xs + 4.f; mr[3] = ys + 4.f; mr[4] = lay;
  }
}

// ---------------------------------------------------------------------------
// NMS (on dets[:,1:5] exactly as the reference does) + keep-select + layer sort
// ---------------------------------------------------------------------------
__global__ __launch_bounds__(512) void nms_select_kernel(
    const float* __restrict__ dets, const float* __restrict__ bws,
    const float* __restrict__ mb, float* __restrict__ detsOut,
    float* __restrict__ mbF, float* __restrict__ nbF,
    int* __restrict__ clsF, int* __restrict__ layF) {
  __shared__ float bx1[KEEP], by1[KEEP], bx2[KEEP], by2[KEEP], sc[KEEP], area[KEEP], vs[KEEP];
  __shared__ int order[KEEP], sel[KEEP], perm[KEEP];
  __shared__ unsigned char supp[KEEP], kp[KEEP];
  __shared__ int s_idx, s_k;
  const int tid = threadIdx.x;

  for (int i = tid; i < KEEP; i += 512) {
    bx1[i] = dets[i * 8 + 1]; by1[i] = dets[i * 8 + 2];
    bx2[i] = dets[i * 8 + 3]; by2[i] = dets[i * 8 + 4];
    sc[i] = dets[i * 8 + 4];
    area[i] = (bx2[i] - bx1[i]) * (by2[i] - by1[i]);
    supp[i] = 0; kp[i] = 0;
  }
  __syncthreads();
  for (int i = tid; i < KEEP; i += 512) {           // order = stable argsort(-sc)
    int r = 0; float v = sc[i];
    for (int q = 0; q < KEEP; ++q) { float o = sc[q]; if (o > v || (o == v && q < i)) r++; }
    order[r] = i;
  }
  __syncthreads();
  for (int r = 0; r < KEEP; ++r) {                  // sequential greedy NMS
    if (tid == 0) {
      int idx = order[r]; int k = supp[idx] ? 0 : 1;
      kp[idx] = (unsigned char)k; s_idx = idx; s_k = k;
    }
    __syncthreads();
    if (s_k) {
      int idx = s_idx;
      for (int j = tid; j < KEEP; j += 512) {
        float ix1 = fmaxf(bx1[idx], bx1[j]), iy1 = fmaxf(by1[idx], by1[j]);
        float ix2 = fminf(bx2[idx], bx2[j]), iy2 = fminf(by2[idx], by2[j]);
        float inter = fmaxf(ix2 - ix1, 0.f) * fmaxf(iy2 - iy1, 0.f);
        float iou = inter / (area[idx] + area[j] - inter + 1e-9f);
        if (iou > IOU_TH) supp[j] = 1;
      }
    }
    __syncthreads();
  }
  for (int i = tid; i < KEEP; i += 512) vs[i] = kp[i] ? sc[i] : -1e9f;
  __syncthreads();
  for (int i = tid; i < KEEP; i += 512) {           // sel = stable argsort(-vs)
    int r = 0; float v = vs[i];
    for (int q = 0; q < KEEP; ++q) { float o = vs[q]; if (o > v || (o == v && q < i)) r++; }
    sel[r] = i;
  }
  __syncthreads();
  for (int s = tid; s < KEEP; s += 512) vs[s] = mb[sel[s] * 5 + 4];
  __syncthreads();
  for (int s = tid; s < KEEP; s += 512) {           // ord2 = stable argsort(-layer)
    int r = 0; float v = vs[s];
    for (int q = 0; q < KEEP; ++q) { float o = vs[q]; if (o > v || (o == v && q < s)) r++; }
    perm[r] = sel[s];
  }
  __syncthreads();
  for (int s = tid; s < KEEP; s += 512) {
    int p = perm[s];
    for (int c = 0; c < 8; ++c) detsOut[s * 8 + c] = dets[p * 8 + c];
    for (int c = 0; c < 5; ++c) mbF[s * 5 + c] = mb[p * 5 + c];
    float m0 = mb[p * 5 + 0], m1 = mb[p * 5 + 1];
    nbF[s * 4 + 0] = SCALE_C * (bws[p * 8 + 0] - m0);
    nbF[s * 4 + 1] = SCALE_C * (bws[p * 8 + 1] - m1);
    nbF[s * 4 + 2] = SCALE_C * (bws[p * 8 + 2] - m0);
    nbF[s * 4 + 3] = SCALE_C * (bws[p * 8 + 3] - m1);
    int ci = (int)dets[p * 8 + 7]; ci = ci < 0 ? 0 : (ci > NUM_CLASSES - 1 ? NUM_CLASSES - 1 : ci);
    clsF[s] = ci;
    int li = (int)mb[p * 5 + 4]; li = li < 0 ? 0 : (li > 2 ? 2 : li);
    layF[s] = li;
  }
}

// ---------------------------------------------------------------------------
// ROI align (selected level only) -> bf16 rois, mask-head input layout [c][pix]
// ---------------------------------------------------------------------------
__global__ __launch_bounds__(256) void roi_align_kernel(
    const float* f0, const float* f1, const float* f2,
    const float* __restrict__ mbF, const int* __restrict__ layF,
    unsigned short* __restrict__ act9, int cb) {
  const int n = blockIdx.x, g = cb + n;
  const int pix = blockIdx.y;           // 0..80
  const int c = threadIdx.x;            // channel
  const int lay = layF[g];
  const float* feat; int H, W;
  if (lay == 0) { feat = f0; H = 64; W = 64; }
  else if (lay == 1) { feat = f1; H = 32; W = 32; }
  else { feat = f2; H = 16; W = 16; }
  float x1 = mbF[g * 5 + 0], y1 = mbF[g * 5 + 1];
  float x2 = mbF[g * 5 + 2], y2 = mbF[g * 5 + 3];
  int xx = pix % ROI_HW, yy = pix / ROI_HW;
  float gx = x1 + (x2 - x1) * ((xx + 0.5f) / (float)ROI_HW);
  float gy = y1 + (y2 - y1) * ((yy + 0.5f) / (float)ROI_HW);
  gx = fminf(fmaxf(gx, 0.f), W - 1.f);
  gy = fminf(fmaxf(gy, 0.f), H - 1.f);
  int x0 = (int)floorf(gx), y0 = (int)floorf(gy);
  float wx = gx - x0, wy = gy - y0;
  int x1i = min(x0 + 1, W - 1), y1i = min(y0 + 1, H - 1);
  const float* fc = feat + (long)c * H * W;
  float v = (1.f - wy) * ((1.f - wx) * fc[y0 * W + x0] + wx * fc[y0 * W + x1i])
          + wy * ((1.f - wx) * fc[y1i * W + x0] + wx * fc[y1i * W + x1i]);
  act9[(long)n * (256 * 81) + (long)c * 81 + pix] = f2bf(v);
}

// ---------------------------------------------------------------------------
// crop_resize: gather class mask, bilinear resample 36x36 -> 36x36
// ---------------------------------------------------------------------------
__global__ __launch_bounds__(128) void crop_resize_kernel(
    const float* __restrict__ maskAll, const float* __restrict__ nbF,
    const int* __restrict__ clsF, float* __restrict__ outMasks, int cb) {
  const int n = blockIdx.x, g = cb + n;
  const int idx = blockIdx.y * 128 + threadIdx.x;
  if (idx >= MASK_OUT * MASK_OUT) return;
  const int ox = idx % MASK_OUT, oy = idx / MASK_OUT;
  const float* m = maskAll + (long)n * NUM_CLASSES * 1296 + (long)clsF[g] * 1296;
  float x1 = nbF[g * 4 + 0], y1 = nbF[g * 4 + 1];
  float x2 = nbF[g * 4 + 2], y2 = nbF[g * 4 + 3];
  float gx = x1 + (x2 - x1) * ((float)ox / 35.f);
  float gy = y1 + (y2 - y1) * ((float)oy / 35.f);
  gx = fminf(fmaxf(gx, 0.f), 35.f);
  gy = fminf(fmaxf(gy, 0.f), 35.f);
  int x0 = (int)floorf(gx), y0 = (int)floorf(gy);
  float wx = gx - x0, wy = gy - y0;
  int x1i = min(x0 + 1, 35), y1i = min(y0 + 1, 35);
  float v = (1.f - wy) * ((1.f - wx) * m[y0 * 36 + x0] + wx * m[y0 * 36 + x1i])
          + wy * ((1.f - wx) * m[y1i * 36 + x0] + wx * m[y1i * 36 + x1i]);
  outMasks[(long)g * 1296 + idx] = v;
}

// ---------------------------------------------------------------------------
// Host launcher
// ---------------------------------------------------------------------------
extern "C" void kernel_launch(void* const* d_in, const int* in_sizes, int n_in,
                              void* d_out, int out_size, void* d_ws, size_t ws_size,
                              hipStream_t stream) {
  (void)in_sizes; (void)n_in; (void)out_size;
  // input order: per-level (heat, tl, br, feat) x3, then weights
  const float* heat0 = (const float*)d_in[0];
  const float* tl0   = (const float*)d_in[1];
  const float* br0   = (const float*)d_in[2];
  const float* feat0 = (const float*)d_in[3];
  const float* heat1 = (const float*)d_in[4];
  const float* tl1   = (const float*)d_in[5];
  const float* br1   = (const float*)d_in[6];
  const float* feat1 = (const float*)d_in[7];
  const float* heat2 = (const float*)d_in[8];
  const float* tl2   = (const float*)d_in[9];
  const float* br2   = (const float*)d_in[10];
  const float* feat2 = (const float*)d_in[11];
  const float* c1_w  = (const float*)d_in[12];
  const float* c1_b  = (const float*)d_in[13];
  const float* mid_w = (const float*)d_in[14];
  const float* mid_b = (const float*)d_in[15];
  const float* bn_g  = (const float*)d_in[16];
  const float* bn_b  = (const float*)d_in[17];
  const float* d1_w  = (const float*)d_in[18];
  const float* d1_b  = (const float*)d_in[19];
  const float* d2_w  = (const float*)d_in[20];
  const float* d2_b  = (const float*)d_in[21];
  const float* c5_w  = (const float*)d_in[22];
  const float* c5_b  = (const float*)d_in[23];
  const float* c6_w  = (const float*)d_in[24];
  const float* c6_b  = (const float*)d_in[25];

  float* out = (float*)d_out;          // [0..2399] dets, [2400..] masks

  // ---- workspace carving -------------------------------------------------
  char* base = (char*)d_ws;
  size_t off = 0;
  auto alloc = [&](size_t bytes) -> void* {
    void* p = base + off;
    off = (off + bytes + 255) & ~(size_t)255;
    return p;
  };
  float* topVal = (float*)alloc(3 * KTOP * 4);
  int*   topIdx = (int*)  alloc(3 * KTOP * 4);
  float* dets   = (float*)alloc(KEEP * 8 * 4);
  float* bwsA   = (float*)alloc(KEEP * 8 * 4);
  float* mbA    = (float*)alloc(KEEP * 5 * 4);
  float* mbF    = (float*)alloc(KEEP * 5 * 4);
  float* nbF    = (float*)alloc(KEEP * 4 * 4);
  int*   clsF   = (int*)  alloc(KEEP * 4);
  int*   layF   = (int*)  alloc(KEEP * 4);
  unsigned short* Ac1  = (unsigned short*)alloc((size_t)589824 * 2);
  unsigned short* Amid = (unsigned short*)alloc((size_t)3 * 589824 * 2);
  unsigned short* Ac5  = (unsigned short*)alloc((size_t)589824 * 2);
  unsigned short* Ad1  = (unsigned short*)alloc((size_t)262144 * 2);
  unsigned short* Ad2  = (unsigned short*)alloc((size_t)262144 * 2);
  unsigned short* Ac6  = (unsigned short*)alloc((size_t)20480 * 2);

  const size_t perRoi = (size_t)2 * 20736 * 2   // act9 a/b (bf16)
                      + (size_t)2 * 82944 * 2   // act18 a/b
                      + (size_t)331776 * 2      // act36
                      + (size_t)103680 * 4;     // maskAll (f32)
  int chunk = 1;
  if (ws_size > off + 4096) {
    size_t c = (ws_size - off - 4096) / perRoi;
    if (c < 1) c = 1; if (c > KEEP) c = KEEP;
    chunk = (int)c;
  }
  unsigned short* act9a  = (unsigned short*)alloc((size_t)chunk * 20736 * 2);
  unsigned short* act9b  = (unsigned short*)alloc((size_t)chunk * 20736 * 2);
  unsigned short* act18a = (unsigned short*)alloc((size_t)chunk * 82944 * 2);
  unsigned short* act18b = (unsigned short*)alloc((size_t)chunk * 82944 * 2);
  unsigned short* act36  = (unsigned short*)alloc((size_t)chunk * 331776 * 2);
  float*          maskAll= (float*)         alloc((size_t)chunk * 103680 * 4);

  // ---- weight pre-pack ---------------------------------------------------
  pk_f32_to_bf16<<<(589824 + 255) / 256, 256, 0, stream>>>(c1_w, Ac1, 589824);
  pk_f32_to_bf16<<<(1769472 + 255) / 256, 256, 0, stream>>>(mid_w, Amid, 1769472);
  pk_f32_to_bf16<<<(589824 + 255) / 256, 256, 0, stream>>>(c5_w, Ac5, 589824);
  pk_f32_to_bf16<<<(20480 + 255) / 256, 256, 0, stream>>>(c6_w, Ac6, 20480);
  pk_deconv<<<(262144 + 255) / 256, 256, 0, stream>>>(d1_w, Ad1);
  pk_deconv<<<(262144 + 255) / 256, 256, 0, stream>>>(d2_w, Ad2);

  // ---- decode / NMS front-end -------------------------------------------
  topk_kernel<<<3, 1024, 0, stream>>>(heat0, heat1, heat2, topVal, topIdx);
  decode_kernel<<<3, 128, 0, stream>>>(tl0, tl1, tl2, br0, br1, br2,
                                       topVal, topIdx, dets, bwsA, mbA);
  nms_select_kernel<<<1, 512, 0, stream>>>(dets, bwsA, mbA, out, mbF, nbF, clsF, layF);

  // ---- mask head descriptors --------------------------------------------
  GemmDesc dc1   = {256, 2304, 256, 9, 9, 9, 9, 3, 3, 1, 81, 0, 1, 0, 20736, 20736};
  GemmDesc ddc1  = {256, 256, 256, 9, 9, 18, 18, 1, 1, 0, 81, 1, 1, 0, 20736, 82944};
  GemmDesc dc5   = {256, 2304, 256, 18, 18, 18, 18, 3, 3, 1, 324, 0, 1, 0, 82944, 82944};
  GemmDesc ddc2  = {256, 256, 256, 18, 18, 36, 36, 1, 1, 0, 324, 1, 1, 0, 82944, 331776};
  GemmDesc dc6   = {80, 256, 256, 36, 36, 36, 36, 1, 1, 0, 1296, 0, 2, 1, 331776, 103680};
  // dynamic LDS for the packed B panel: K * 32 bytes
  const size_t ldsK2304 = (size_t)2304 * 32;   // 73728 B (3x3 layers)
  const size_t ldsK256  = (size_t)256 * 32;    // 8192 B  (deconv / 1x1 layers)

  for (int cb = 0; cb < KEEP; cb += chunk) {
    int cn = (KEEP - cb < chunk) ? (KEEP - cb) : chunk;
    roi_align_kernel<<<dim3(cn, 81), 256, 0, stream>>>(feat0, feat1, feat2,
                                                       mbF, layF, act9a, cb);
    // c1 + BN0 + relu
    wmma_gemm_kernel<<<dim3(cn, 6, 1), 256, ldsK2304, stream>>>(
        Ac1, act9a, act9b, c1_b, bn_g + 0 * 256, bn_b + 0 * 256, dc1);
    // mid 0..2 (ping-pong b->a->b->a)
    wmma_gemm_kernel<<<dim3(cn, 6, 1), 256, ldsK2304, stream>>>(
        Amid + 0 * 589824, act9b, act9a, mid_b + 0 * 256, bn_g + 1 * 256, bn_b + 1 * 256, dc1);
    wmma_gemm_kernel<<<dim3(cn, 6, 1), 256, ldsK2304, stream>>>(
        Amid + 1 * 589824, act9a, act9b, mid_b + 1 * 256, bn_g + 2 * 256, bn_b + 2 * 256, dc1);
    wmma_gemm_kernel<<<dim3(cn, 6, 1), 256, ldsK2304, stream>>>(
        Amid + 2 * 589824, act9b, act9a, mid_b + 2 * 256, bn_g + 3 * 256, bn_b + 3 * 256, dc1);
    // deconv1 (4 parity classes) + relu
    wmma_gemm_kernel<<<dim3(cn, 6, 4), 256, ldsK256, stream>>>(
        Ad1, act9a, act18a, d1_b, nullptr, nullptr, ddc1);
    // c5 + BN4 + relu
    wmma_gemm_kernel<<<dim3(cn, 21, 1), 256, ldsK2304, stream>>>(
        Ac5, act18a, act18b, c5_b, bn_g + 4 * 256, bn_b + 4 * 256, dc5);
    // deconv2 (4 parity classes) + relu
    wmma_gemm_kernel<<<dim3(cn, 21, 4), 256, ldsK256, stream>>>(
        Ad2, act18b, act36, d2_b, nullptr, nullptr, ddc2);
    // c6 (1x1, 80 classes) + sigmoid -> f32 masks
    wmma_gemm_kernel<<<dim3(cn, 81, 1), 256, ldsK256, stream>>>(
        Ac6, act36, maskAll, c6_b, nullptr, nullptr, dc6);
    // final crop+resize into output
    crop_resize_kernel<<<dim3(cn, (1296 + 127) / 128), 128, 0, stream>>>(
        maskAll, nbF, clsF, out + 2400, cb);
  }
}
